// Memory_Network_27848567947821
// MI455X (gfx1250) — matrix operand order repaired and checked by reading, not compiled, not run on hardware
//
#include <hip/hip_runtime.h>
#include <hip/hip_bf16.h>
#include <math.h>

typedef __attribute__((ext_vector_type(16))) _Float16 v16h;
typedef __attribute__((ext_vector_type(8)))  float    v8f;

constexpr int Cc    = 64;       // channels
constexpr int Nn    = 20;       // memory slots
constexpr int NPAD  = 32;       // padded n for WMMA K/N dims
constexpr int Kc    = 20;       // n_class + 1
constexpr int PLANE = 65536;    // h*w (256*256)
constexpr int Bb    = 4;        // batch
constexpr int TOTAL = Bb * PLANE;             // 262144 pixels
constexpr int MAX_TILES  = TOTAL / 16 + 32;   // 16416 (covers per-class pad)
constexpr int MAX_PADDED = MAX_TILES * 16;
constexpr int FRAG = 32 * 16;                 // halves per fragment (512)

#define EPSF 1e-12f

// ---------------- init: zero counters, poison bin index array -------------
__global__ void init_kernel(int* counts, int* cursor, int* binIdx) {
    int i = blockIdx.x * blockDim.x + threadIdx.x;
    if (i < 32) { counts[i] = 0; cursor[i] = 0; }
    for (int j = i; j < MAX_PADDED; j += gridDim.x * blockDim.x) binIdx[j] = -1;
}

// ------- prep: keys/values pre-swizzled into WMMA fragment layout ---------
// knfrag[k][s][nt][L][hi]  : B-frag for scores matmul   (k: class, s: K-step)
//   element = Khat[c = 32*s + 8*(L>>4) + (hi&7) + 16*(hi>>3)][n = nt*16 + (L&15)]
// valfrag[k][ct][L][hi]    : B-frag for output matmul   (ct: 16-col tile)
//   element = V[n = 8*(L>>4) + (hi&7) + 16*(hi>>3)][c = ct*16 + (L&15)]
__global__ __launch_bounds__(256)
void prep_mem_kernel(const float* __restrict__ key_mem,
                     const float* __restrict__ val_mem,
                     _Float16* __restrict__ knfrag,
                     _Float16* __restrict__ valfrag,
                     float* __restrict__ bias) {
    int tid = threadIdx.x;
    __shared__ float invn[Kc * Nn];
    for (int col = tid; col < Kc * Nn; col += blockDim.x) {
        int k = col / Nn, n = col % Nn;
        float ssq = 0.f;
        for (int c = 0; c < Cc; ++c) {
            float v = key_mem[(k * Cc + c) * Nn + n];
            ssq += v * v;
        }
        invn[col] = 1.0f / fmaxf(sqrtf(ssq), EPSF);
    }
    __syncthreads();
    // knfrag: i = k*2048 + s*1024 + nt*512 + L*16 + hi
    for (int i = tid; i < Kc * 4 * FRAG; i += blockDim.x) {
        int hi = i & 15, L = (i >> 4) & 31, nt = (i >> 9) & 1, s = (i >> 10) & 1;
        int k = i >> 11;
        int n = nt * 16 + (L & 15);
        int c = 32 * s + 8 * (L >> 4) + (hi & 7) + 16 * (hi >> 3);
        float v = (n < Nn) ? key_mem[(k * Cc + c) * Nn + n] * invn[k * Nn + n] : 0.f;
        knfrag[i] = (_Float16)v;
    }
    // valfrag: i = k*2048 + ct*512 + L*16 + hi
    for (int i = tid; i < Kc * 4 * FRAG; i += blockDim.x) {
        int hi = i & 15, L = (i >> 4) & 31, ct = (i >> 9) & 3;
        int k = i >> 11;
        int n = 8 * (L >> 4) + (hi & 7) + 16 * (hi >> 3);
        int c = ct * 16 + (L & 15);
        float v = (n < Nn) ? val_mem[(k * Nn + n) * Cc + c] : 0.f;
        valfrag[i] = (_Float16)v;
    }
    // bias[k][c] = sum_{k'} mean_n V[k'][:,c] - mean_n V[k][:,c]
    __shared__ float colsum[Kc * Cc];
    for (int i = tid; i < Kc * Cc; i += blockDim.x) {
        int k = i / Cc, c = i % Cc;
        float s = 0.f;
        for (int n = 0; n < Nn; ++n) s += val_mem[(k * Nn + n) * Cc + c];
        colsum[i] = s;
    }
    __syncthreads();
    for (int i = tid; i < Kc * Cc; i += blockDim.x) {
        int c = i % Cc;
        float T = 0.f;
        for (int kk = 0; kk < Kc; ++kk) T += colsum[kk * Cc + c];
        bias[i] = (T - colsum[i]) * (1.0f / (float)Nn);
    }
}

// ---------------- histogram of classes ------------------------------------
__global__ void count_kernel(const int* __restrict__ seg, int* counts) {
    int i = blockIdx.x * blockDim.x + threadIdx.x;
    if (i < TOTAL) atomicAdd(&counts[seg[i] + 1], 1);
}

// ---------------- 16-aligned exclusive scan -------------------------------
__global__ void scan_kernel(const int* __restrict__ counts, int* offsets,
                            int* cursor, int* numTiles) {
    if (threadIdx.x == 0) {
        int off = 0;
        for (int k = 0; k < Kc; ++k) {
            offsets[k] = off;
            cursor[k]  = off;
            off += (counts[k] + 15) & ~15;
        }
        offsets[Kc] = off;
        *numTiles = off >> 4;
    }
}

// ---------------- tile -> class map ---------------------------------------
__global__ void tileclass_kernel(const int* __restrict__ offsets,
                                 const int* __restrict__ numTiles,
                                 int* __restrict__ tileClass) {
    int t = blockIdx.x * blockDim.x + threadIdx.x;
    if (t < *numTiles) {
        int start = t << 4;
        int k = 0;
        for (int kk = 0; kk < Kc; ++kk)
            if (start >= offsets[kk]) k = kk;
        tileClass[t] = k;
    }
}

// ---------------- scatter pixel indices into class bins -------------------
__global__ void scatter_kernel(const int* __restrict__ seg, int* cursor,
                               int* __restrict__ binIdx) {
    int i = blockIdx.x * blockDim.x + threadIdx.x;
    if (i < TOTAL) {
        int k = seg[i] + 1;
        int slot = atomicAdd(&cursor[k], 1);
        binIdx[slot] = i;
    }
}

// ---------------- main WMMA kernel: one wave per 16-pixel tile ------------
__global__ __launch_bounds__(32)
void attn_wmma_kernel(const float* __restrict__ content,
                      const int* __restrict__ binIdx,
                      const int* __restrict__ tileClass,
                      const int* __restrict__ numTiles,
                      const _Float16* __restrict__ knfrag,
                      const _Float16* __restrict__ valfrag,
                      const float* __restrict__ bias,
                      float* __restrict__ out) {
    int t = blockIdx.x;
    if (t >= *numTiles) return;               // wave-uniform exit
    int k = tileClass[t];
    int lane = threadIdx.x;
    int laneLo = lane & 15;
    int laneHi = lane >> 4;

    __shared__ int idxSh[16];
    __shared__ __align__(32) _Float16 fragA[2][32][16];  // A-frags, per K-step
    __shared__ __align__(32) float    S[16][NPAD];       // raw scores
    __shared__ __align__(32) _Float16 fragW[32][16];     // attn-weight A-frag

    if (lane < 16) idxSh[lane] = binIdx[t * 16 + lane];
    __syncthreads();

    // ---- gather + L2-normalize; write f16 A-fragments (packed pairs) -----
    int r = laneLo;
    int cbase = laneHi * 32;
    int idx = idxSh[r];
    int batch = (idx >= 0) ? (idx >> 16) : 0;
    int p     = (idx >= 0) ? (idx & (PLANE - 1)) : 0;
    const float* basePtr = content + (size_t)batch * Cc * PLANE + p;

    float vbuf[32];
    float ss = 0.f;
#pragma unroll
    for (int j = 0; j < 32; ++j) {
        float v = (idx >= 0) ? basePtr[(size_t)(cbase + j) * PLANE] : 0.f;
        vbuf[j] = v;
        ss += v * v;
    }
    ss += __shfl_xor(ss, 16, 32);             // combine the two channel halves
    float inv = 1.0f / fmaxf(sqrtf(ss), EPSF);
#pragma unroll
    for (int j = 0; j < 32; j += 2) {
        int c0 = cbase + j;
        int s  = c0 >> 5;                     // == laneHi, uniform per lane
        int kl = c0 & 31;
        int lh = (kl >> 3) & 1;               // target lane half
        int hi = (kl & 7) + 8 * (kl >> 4);    // target half index (even)
        auto pk = __builtin_amdgcn_cvt_pkrtz(vbuf[j] * inv, vbuf[j + 1] * inv);
        *(decltype(pk)*)&fragA[s][lh * 16 + r][hi] = pk;
    }
    __syncthreads();

    // ---- scores = Ahat(16x64) @ Khat[k](64x32pad): 2 K-steps x 2 N-tiles --
    const _Float16* kf = knfrag + (size_t)k * 4 * FRAG;
    v8f acc0 = {}, acc1 = {};
#pragma unroll
    for (int s = 0; s < 2; ++s) {
        v16h a  = *(const v16h*)&fragA[s][lane][0];
        v16h b0 = *(const v16h*)&kf[(s * 2 + 0) * FRAG + lane * 16];
        v16h b1 = *(const v16h*)&kf[(s * 2 + 1) * FRAG + lane * 16];
        acc0 = __builtin_amdgcn_wmma_f32_16x16x32_f16(false, a, false, b0,
                                                      (short)0, acc0, false, false);
        acc1 = __builtin_amdgcn_wmma_f32_16x16x32_f16(false, a, false, b1,
                                                      (short)0, acc1, false, false);
    }
#pragma unroll
    for (int i = 0; i < 8; ++i) {             // C layout: VGPR i -> rows i / i+8
        S[i + 8 * laneHi][laneLo]      = acc0[i];
        S[i + 8 * laneHi][16 + laneLo] = acc1[i];
    }
    __syncthreads();

    // ---- softmax over n=20; write packed f16 weight fragment --------------
    if (lane < 16) {
        float mx = -1e30f;
#pragma unroll
        for (int n = 0; n < Nn; ++n) mx = fmaxf(mx, S[lane][n]);
        float e[Nn];
        float sum = 0.f;
#pragma unroll
        for (int n = 0; n < Nn; ++n) {
            e[n] = __expf(S[lane][n] - mx);
            sum += e[n];
        }
        float is = 1.0f / sum;
#pragma unroll
        for (int kl = 0; kl < NPAD; kl += 2) {
            float w0 = (kl     < Nn) ? e[kl]     * is : 0.f;
            float w1 = (kl + 1 < Nn) ? e[kl + 1] * is : 0.f;
            int lh = (kl >> 3) & 1;
            int hi = (kl & 7) + 8 * (kl >> 4);
            auto pk = __builtin_amdgcn_cvt_pkrtz(w0, w1);
            *(decltype(pk)*)&fragW[lh * 16 + lane][hi] = pk;
        }
    }
    __syncthreads();

    // ---- out = attn(16x32pad) @ V[k](32x64) + bias[k]: 4 N-tiles ----------
    const _Float16* vf = valfrag + (size_t)k * 4 * FRAG;
    v16h a2 = *(const v16h*)&fragW[lane][0];
#pragma unroll
    for (int ct = 0; ct < 4; ++ct) {
        int ccol = ct * 16 + laneLo;
        float bv = bias[k * Cc + ccol];
        v8f acc = { bv, bv, bv, bv, bv, bv, bv, bv };
        v16h b = *(const v16h*)&vf[ct * FRAG + lane * 16];
        acc = __builtin_amdgcn_wmma_f32_16x16x32_f16(false, a2, false, b,
                                                     (short)0, acc, false, false);
#pragma unroll
        for (int i = 0; i < 8; ++i) {
            int idx2 = idxSh[i + 8 * laneHi];
            if (idx2 >= 0) {
                int bb = idx2 >> 16;
                int pp = idx2 & (PLANE - 1);
                out[(size_t)(bb * Cc + ccol) * PLANE + pp] = acc[i];
            }
        }
    }
}

// ---------------------------------------------------------------------------
extern "C" void kernel_launch(void* const* d_in, const int* in_sizes, int n_in,
                              void* d_out, int out_size, void* d_ws, size_t ws_size,
                              hipStream_t stream) {
    const float* content = (const float*)d_in[0];
    const int*   seg     = (const int*)d_in[1];
    const float* key_mem = (const float*)d_in[2];
    const float* val_mem = (const float*)d_in[3];
    float* out = (float*)d_out;

    // workspace carving (fragment arrays 32B-aligned: 279200*4 % 32 == 0)
    int* counts    = (int*)d_ws;
    int* cursor    = counts + 32;
    int* offsets   = cursor + 32;
    int* numTiles  = offsets + 32;
    int* tileClass = numTiles + 32;
    int* binIdx    = tileClass + MAX_TILES;
    _Float16* knfrag  = (_Float16*)(binIdx + MAX_PADDED);
    _Float16* valfrag = knfrag + (size_t)Kc * 4 * FRAG;
    float* bias       = (float*)(valfrag + (size_t)Kc * 4 * FRAG);

    init_kernel<<<(MAX_PADDED + 255) / 256, 256, 0, stream>>>(counts, cursor, binIdx);
    prep_mem_kernel<<<1, 256, 0, stream>>>(key_mem, val_mem, knfrag, valfrag, bias);
    count_kernel<<<(TOTAL + 255) / 256, 256, 0, stream>>>(seg, counts);
    scan_kernel<<<1, 32, 0, stream>>>(counts, offsets, cursor, numTiles);
    tileclass_kernel<<<(MAX_TILES + 255) / 256, 256, 0, stream>>>(offsets, numTiles, tileClass);
    scatter_kernel<<<(TOTAL + 255) / 256, 256, 0, stream>>>(seg, cursor, binIdx);
    attn_wmma_kernel<<<MAX_TILES, 32, 0, stream>>>(content, binIdx, tileClass, numTiles,
                                                   knfrag, valfrag, bias, out);
}
